// TriangleDistToZ_73083163509156
// MI455X (gfx1250) — matrix operand order repaired and checked by reading, not compiled, not run on hardware
//
#include <hip/hip_runtime.h>
#include <hip/hip_bf16.h>

#define E 256
#define C 128
#define NP 384
#define NL 96
#define BB 2
#define EPS 1e-5f

typedef _Float16 h16;
typedef h16  v16h __attribute__((ext_vector_type(16)));
typedef float v8f  __attribute__((ext_vector_type(8)));

__device__ __forceinline__ v8f wmma16(v16h a, v16h b, v8f c) {
  // D = A(16x32 f16) x B(32x16 f16) + C(f32)
  return __builtin_amdgcn_wmma_f32_16x16x32_f16(false, a, false, b, (short)0, c, false, false);
}

// fast sigmoid: v_exp_f32 + v_rcp_f32 (both TRANS -> co-execute with WMMA)
__device__ __forceinline__ float fsigmoid(float x) {
  return __builtin_amdgcn_rcpf(1.f + __expf(-x));
}

// A-fragment slot (16x32 f16, contiguous k = 16*(lane/16) + h, this code's order): element (m 0..15, k 0..31) -> (lane, half index 0..15)
__device__ __forceinline__ void a_slot(int m, int k, int& lane, int& h) {
  lane = m + ((k >> 4) << 4);
  h = k & 15;
}
// B-fragment slot (32x16 f16): element (k 0..31, n 0..15) -> (lane, half index)
__device__ __forceinline__ void b_slot(int k, int n, int& lane, int& h) {
  lane = n + ((k >> 4) << 4);
  h = k & 15;
}

// ---------------- weight packing into B-fragment panels -----------------
__global__ __launch_bounds__(256) void k_pack(const float* __restrict__ src,
                                              h16* __restrict__ dst, int K, int N) {
  int idx = blockIdx.x * 256 + threadIdx.x;
  if (idx >= K * N) return;
  int k = idx / N, n = idx % N;
  int kt = k >> 5, kin = k & 31, nt = n >> 4, nin = n & 15;
  int lane, h; b_slot(kin, nin, lane, h);
  int KT = K >> 5;
  dst[((nt * KT + kt) * 32 + lane) * 16 + h] = (h16)src[k * N + n];
}

// ---------------- fused LayerNorm + gated projections (WMMA) ------------
struct ProjArgs {
  const float* in;   const float* mask;
  const float* ln_w; const float* ln_b;
  const h16 *pG1, *pV1, *pG2, *pV2, *pG3, *pV3, *pGe;
  const float *gb1, *b1, *gb2, *b2, *gb3, *b3, *gbe;
  h16 *o1, *o2, *o3, *og;
  int mode; // 0=z(ab1T,ab2T,ab3,g) 1=protein_pair(ppT) 2=l_pair(lpT,swapped) 3=inter_pair(ip)
};

__global__ __launch_bounds__(256) void k_ln_gated(ProjArgs A) {
  __shared__ __align__(32) h16 aP[8][32][16];     // A fragments, 8 K-tiles of 32 (K=E=256)
  __shared__ float red[16][17], red2[16][17];
  __shared__ float s_mu[16], s_rs[16], s_mask[16];
  __shared__ int   s_rowbase[16];                  // per-row base for transposed outputs
  const int tid = threadIdx.x;
  const int row = tid >> 4, cg = tid & 15;
  const int pos0 = blockIdx.x * 16;

  // ---- LayerNorm of 16 rows x 256 cols ----
  float x[16];
  const float* src = A.in + (size_t)(pos0 + row) * E + cg * 16;
  float s = 0.f, ss = 0.f;
#pragma unroll
  for (int e = 0; e < 16; ++e) { float v = src[e]; x[e] = v; s += v; ss += v * v; }
  red[row][cg] = s; red2[row][cg] = ss;
  __syncthreads();
  if (cg == 0) {
    float ts = 0.f, tss = 0.f;
#pragma unroll
    for (int e = 0; e < 16; ++e) { ts += red[row][e]; tss += red2[row][e]; }
    float mu = ts * (1.f / E);
    float var = tss * (1.f / E) - mu * mu;
    s_mu[row] = mu; s_rs[row] = rsqrtf(var + EPS);
    s_mask[row] = A.mask ? A.mask[pos0 + row] : 1.f;
    // per-row output base (b,i,j decomposition done once per row, 32-bit)
    int posI = pos0 + row, rb = 0;
    if (A.mode == 0) {                 // ab1T/ab2T: [b][c][Np][Nl]
      int b = posI / (NP * NL), rem = posI % (NP * NL);
      rb = b * (C * NP * NL) + rem;    // rem = i*NL + j already
    } else if (A.mode == 1) {          // ppT: [b][c][Np][Np]
      int b = posI / (NP * NP), rem = posI % (NP * NP);
      rb = b * (C * NP * NP) + rem;    // rem = i*NP + k
    } else if (A.mode == 2) {          // lpT: [b][c][k][j] = lp[b,j,k,c]
      int b = posI / (NL * NL), rem = posI % (NL * NL);
      int j = rem / NL, k = rem % NL;
      rb = b * (C * NL * NL) + k * NL + j;
    }
    s_rowbase[row] = rb;
  }
  __syncthreads();
  float mu = s_mu[row], rs = s_rs[row];
#pragma unroll
  for (int e = 0; e < 16; ++e) {
    int c = cg * 16 + e;
    float v = (x[e] - mu) * rs * A.ln_w[c] + A.ln_b[c];
    int lane, h; a_slot(row, c & 31, lane, h);
    aP[c >> 5][lane][h] = (h16)v;
  }
  __syncthreads();

  // ---- WMMA projections ----
  const int wave = tid >> 5, lane = tid & 31;
  const int nl = lane & 15, mb = (lane >> 4) * 8;
  const int njobs = (A.mode == 0) ? 40 : 8;  // z: 3 gated pairs x8 coltiles + 16 gWe tiles
  for (int job = wave; job < njobs; job += 8) {
    if (A.mode != 0 || job < 24) {
      int p, ct;
      if (A.mode == 0) { p = job >> 3; ct = job & 7; }
      else             { ct = job; p = (A.mode == 1) ? 1 : 0; }
      const h16 *pG, *pV; const float *gB, *vB;
      if (p == 0)      { pG = A.pG1; pV = A.pV1; gB = A.gb1; vB = A.b1; }
      else if (p == 1) { pG = A.pG2; pV = A.pV2; gB = A.gb2; vB = A.b2; }
      else             { pG = A.pG3; pV = A.pV3; gB = A.gb3; vB = A.b3; }
      v8f accG = {0.f,0.f,0.f,0.f,0.f,0.f,0.f,0.f};
      v8f accV = {0.f,0.f,0.f,0.f,0.f,0.f,0.f,0.f};
#pragma unroll
      for (int kt = 0; kt < 8; ++kt) {
        v16h a  = *(const v16h*)&aP[kt][lane][0];
        v16h bg = *(const v16h*)(pG + ((ct * 8 + kt) * 32 + lane) * 16);
        v16h bv = *(const v16h*)(pV + ((ct * 8 + kt) * 32 + lane) * 16);
        accG = wmma16(a, bg, accG);
        accV = wmma16(a, bv, accV);
      }
      int c = ct * 16 + nl;
      float gb = gB[c], vb = vB[c];
      int cstride = (A.mode == 1) ? (NP * NP) : (A.mode == 2 ? (NL * NL) : (NP * NL));
#pragma unroll
      for (int r = 0; r < 8; ++r) {
        int m = mb + r;
        int posI = pos0 + m;
        float gate = fsigmoid(accG[r] + gb);
        float val  = (accV[r] + vb) * gate * s_mask[m];
        if (A.mode == 0) {
          if (p == 2) A.o3[posI * C + c] = (h16)val;                  // ab3 channel-last
          else {
            h16* o = (p == 0) ? A.o1 : A.o2;                          // ab1T / ab2T
            o[s_rowbase[m] + c * (NP * NL)] = (h16)val;
          }
        } else if (A.mode == 3) {
          A.o1[posI * C + c] = (h16)val;                              // ip channel-last
        } else {
          A.o1[s_rowbase[m] + c * cstride] = (h16)val;                // ppT / lpT
        }
      }
    } else {  // ending gate g = sigmoid(zn @ gWe + gbe), E->E
      int ct = job - 24;
      v8f acc = {0.f,0.f,0.f,0.f,0.f,0.f,0.f,0.f};
#pragma unroll
      for (int kt = 0; kt < 8; ++kt) {
        v16h a  = *(const v16h*)&aP[kt][lane][0];
        v16h bg = *(const v16h*)(A.pGe + ((ct * 8 + kt) * 32 + lane) * 16);
        acc = wmma16(a, bg, acc);
      }
      int c = ct * 16 + nl;
      float gb = A.gbe[c];
#pragma unroll
      for (int r = 0; r < 8; ++r) {
        int posI = pos0 + mb + r;
        A.og[posI * E + c] = (h16)fsigmoid(acc[r] + gb);
      }
    }
  }
}

// ---------------- batched per-channel triangle GEMMs --------------------
// out_block[b,i,j,c] (+)= A[batch] (MxK, row-major) @ B[batch] (KxN, row-major)
__global__ __launch_bounds__(256) void k_bgemm(
    const h16* __restrict__ Ab, const h16* __restrict__ Bb, float* __restrict__ outBlock,
    int M, int N, int K, int lda, int ldb,
    int strideA, int strideB, int accumulate, int gm, int gn) {
  __shared__ __align__(32) h16 aT[8][32][16];  // 128x32 A block in fragment order
  __shared__ __align__(32) h16 bT[2][32][16];  // 32x32 B block in fragment order
  int bid = blockIdx.x;
  int batch = bid / (gm * gn); int t = bid % (gm * gn);
  int bm = t / gn, bn = t % gn;
  int b = batch / C, c = batch % C;
  const h16* A  = Ab + (size_t)batch * strideA;
  const h16* Bm = Bb + (size_t)batch * strideB;
  float* outP = outBlock + (size_t)b * (NP * NL * C) + c;
  int m0 = bm * 128, n0 = bn * 32;
  const int tid = threadIdx.x, wave = tid >> 5, lane = tid & 31;
  const int nl = lane & 15, mbase = (lane >> 4) * 8;
  v8f acc0 = {0.f,0.f,0.f,0.f,0.f,0.f,0.f,0.f};
  v8f acc1 = {0.f,0.f,0.f,0.f,0.f,0.f,0.f,0.f};
  const int KS = K >> 5;
  const int ar = tid >> 1, ah = tid & 1;          // A: row 0..127, 16-col half
  const int bkr = tid >> 3, bne = (tid & 7) * 4;  // B: k-row 0..31, 4 cols
  for (int kt = 0; kt < KS; ++kt) {
    { // stage A 128x32 -> fragment panels
      const h16* sA = A + (m0 + ar) * lda + kt * 32 + ah * 16;
      if (kt + 1 < KS) __builtin_prefetch(sA + 32, 0, 1);   // global_prefetch_b8
      int panel = ar >> 4, m = ar & 15;
#pragma unroll
      for (int e = 0; e < 16; ++e) {
        int k = ah * 16 + e;
        int l2, h2; a_slot(m, k, l2, h2);
        aT[panel][l2][h2] = sA[e];
      }
    }
    { // stage B 32x32 -> fragment panels
      const h16* sB = Bm + (kt * 32 + bkr) * ldb + n0 + bne;
      if (kt + 1 < KS) __builtin_prefetch(sB + 32 * ldb, 0, 1);
#pragma unroll
      for (int e = 0; e < 4; ++e) {
        int nloc = bne + e;
        int l2 = (nloc & 15) + ((bkr >> 4) << 4);
        bT[nloc >> 4][l2][bkr & 15] = sB[e];
      }
    }
    __syncthreads();
    v16h a  = *(const v16h*)&aT[wave][lane][0];
    v16h b0 = *(const v16h*)&bT[0][lane][0];
    v16h b1 = *(const v16h*)&bT[1][lane][0];
    acc0 = wmma16(a, b0, acc0);
    acc1 = wmma16(a, b1, acc1);
    __syncthreads();
  }
#pragma unroll
  for (int r = 0; r < 8; ++r) {
    int i = m0 + wave * 16 + mbase + r;
    int a0 = (i * NL + (n0 + nl)) * C;
    int a1 = (i * NL + (n0 + 16 + nl)) * C;
    if (accumulate) { outP[a0] += acc0[r]; outP[a1] += acc1[r]; }
    else            { outP[a0]  = acc0[r]; outP[a1]  = acc1[r]; }
  }
}

// ---------------- final: block+ab3*ip, LN(C), @Ws + bs, gate*mask -------
__global__ __launch_bounds__(256) void k_final(
    const float* __restrict__ blockIn, const h16* __restrict__ ab3,
    const h16* __restrict__ ipb, const h16* __restrict__ gbuf,
    const float* __restrict__ z_mask,
    const float* __restrict__ lnc_w, const float* __restrict__ lnc_b,
    const h16* __restrict__ pWs, const float* __restrict__ bs,
    float* __restrict__ out) {
  __shared__ __align__(32) h16 aP[4][32][16];    // K = C = 128 -> 4 K-tiles
  __shared__ float red[16][17], red2[16][17];
  __shared__ float s_mu[16], s_rs[16], s_mask[16];
  const int tid = threadIdx.x;
  const int row = tid >> 4, cg = tid & 15;
  const int pos0 = blockIdx.x * 16;
  float x[8];
  const int base = (pos0 + row) * C + cg * 8;
  float s = 0.f, ss = 0.f;
#pragma unroll
  for (int e = 0; e < 8; ++e) {
    float v = blockIn[base + e] + (float)ab3[base + e] * (float)ipb[base + e];
    x[e] = v; s += v; ss += v * v;
  }
  red[row][cg] = s; red2[row][cg] = ss;
  __syncthreads();
  if (cg == 0) {
    float ts = 0.f, tss = 0.f;
#pragma unroll
    for (int e = 0; e < 16; ++e) { ts += red[row][e]; tss += red2[row][e]; }
    float mu = ts * (1.f / C); float var = tss * (1.f / C) - mu * mu;
    s_mu[row] = mu; s_rs[row] = rsqrtf(var + EPS);
    s_mask[row] = z_mask[pos0 + row];
  }
  __syncthreads();
  float mu = s_mu[row], rs = s_rs[row];
#pragma unroll
  for (int e = 0; e < 8; ++e) {
    int c = cg * 8 + e;
    float v = (x[e] - mu) * rs * lnc_w[c] + lnc_b[c];
    int lane, h; a_slot(row, c & 31, lane, h);
    aP[c >> 5][lane][h] = (h16)v;
  }
  __syncthreads();
  const int wave = tid >> 5, lane = tid & 31, nl = lane & 15, mb = (lane >> 4) * 8;
  for (int nt = wave; nt < 16; nt += 8) {   // E/16 = 16 output tiles
    v8f acc = {0.f,0.f,0.f,0.f,0.f,0.f,0.f,0.f};
#pragma unroll
    for (int kt = 0; kt < 4; ++kt) {
      v16h a   = *(const v16h*)&aP[kt][lane][0];
      v16h bfr = *(const v16h*)(pWs + ((nt * 4 + kt) * 32 + lane) * 16);
      acc = wmma16(a, bfr, acc);
    }
    int co = nt * 16 + nl;
    float bsv = bs[co];
#pragma unroll
    for (int r = 0; r < 8; ++r) {
      int m = mb + r;
      int posI = pos0 + m;
      float gv = (float)gbuf[posI * E + co];
      out[posI * E + co] = gv * (acc[r] + bsv) * s_mask[m];
    }
  }
}

// ------------------------------ launcher --------------------------------
extern "C" void kernel_launch(void* const* d_in, const int* in_sizes, int n_in,
                              void* d_out, int out_size, void* d_ws, size_t ws_size,
                              hipStream_t stream) {
  (void)in_sizes; (void)n_in; (void)out_size; (void)ws_size;
  const float* z     = (const float*)d_in[0];
  const float* l_p   = (const float*)d_in[1];
  const float* p_p   = (const float*)d_in[2];
  const float* i_p   = (const float*)d_in[3];
  const float* zm    = (const float*)d_in[4];
  const float* ln_w  = (const float*)d_in[5];
  const float* ln_b  = (const float*)d_in[6];
  const float* lnc_w = (const float*)d_in[7];
  const float* lnc_b = (const float*)d_in[8];
  const float* gW1 = (const float*)d_in[9];  const float* gb1 = (const float*)d_in[10];
  const float* gW2 = (const float*)d_in[11]; const float* gb2 = (const float*)d_in[12];
  const float* gW3 = (const float*)d_in[13]; const float* gb3 = (const float*)d_in[14];
  const float* W1  = (const float*)d_in[15]; const float* b1  = (const float*)d_in[16];
  const float* W2  = (const float*)d_in[17]; const float* b2  = (const float*)d_in[18];
  const float* W3  = (const float*)d_in[19]; const float* b3  = (const float*)d_in[20];
  const float* gWe = (const float*)d_in[21]; const float* gbe = (const float*)d_in[22];
  const float* Wsm = (const float*)d_in[23]; const float* bs  = (const float*)d_in[24];
  float* out = (float*)d_out;

  char* w = (char*)d_ws;
  size_t off = 0;
  auto take = [&](size_t bytes) -> char* {
    char* p = w + off; off = (off + bytes + 255) & ~(size_t)255; return p;
  };
  h16* pGW1 = (h16*)take((size_t)E * C * 2);
  h16* pW1p = (h16*)take((size_t)E * C * 2);
  h16* pGW2 = (h16*)take((size_t)E * C * 2);
  h16* pW2p = (h16*)take((size_t)E * C * 2);
  h16* pGW3 = (h16*)take((size_t)E * C * 2);
  h16* pW3p = (h16*)take((size_t)E * C * 2);
  h16* pGWe = (h16*)take((size_t)E * E * 2);
  h16* pWs  = (h16*)take((size_t)C * E * 2);
  h16* ab1T = (h16*)take((size_t)BB * C * NP * NL * 2);
  h16* ab2T = (h16*)take((size_t)BB * C * NP * NL * 2);
  h16* ab3  = (h16*)take((size_t)BB * NP * NL * C * 2);
  h16* ipb  = (h16*)take((size_t)BB * NP * NL * C * 2);
  h16* ppT  = (h16*)take((size_t)BB * C * NP * NP * 2);
  h16* lpT  = (h16*)take((size_t)BB * C * NL * NL * 2);
  h16* gbuf = (h16*)take((size_t)BB * NP * NL * E * 2);
  float* blk = (float*)take((size_t)BB * NP * NL * C * 4);

  // 1) pack weights into WMMA B-fragment panels
  k_pack<<<(E * C + 255) / 256, 256, 0, stream>>>(gW1, pGW1, E, C);
  k_pack<<<(E * C + 255) / 256, 256, 0, stream>>>(W1,  pW1p, E, C);
  k_pack<<<(E * C + 255) / 256, 256, 0, stream>>>(gW2, pGW2, E, C);
  k_pack<<<(E * C + 255) / 256, 256, 0, stream>>>(W2,  pW2p, E, C);
  k_pack<<<(E * C + 255) / 256, 256, 0, stream>>>(gW3, pGW3, E, C);
  k_pack<<<(E * C + 255) / 256, 256, 0, stream>>>(W3,  pW3p, E, C);
  k_pack<<<(E * E + 255) / 256, 256, 0, stream>>>(gWe, pGWe, E, E);
  k_pack<<<(C * E + 255) / 256, 256, 0, stream>>>(Wsm, pWs,  C, E);

  // 2) LN + gated projections
  ProjArgs pa{};
  pa.ln_w = ln_w; pa.ln_b = ln_b;
  pa.pG1 = pGW1; pa.pV1 = pW1p; pa.pG2 = pGW2; pa.pV2 = pW2p;
  pa.pG3 = pGW3; pa.pV3 = pW3p; pa.pGe = pGWe;
  pa.gb1 = gb1; pa.b1 = b1; pa.gb2 = gb2; pa.b2 = b2; pa.gb3 = gb3; pa.b3 = b3; pa.gbe = gbe;

  ProjArgs a1 = pa; a1.in = p_p; a1.mask = nullptr; a1.o1 = ppT; a1.mode = 1;
  k_ln_gated<<<(BB * NP * NP) / 16, 256, 0, stream>>>(a1);
  ProjArgs a2 = pa; a2.in = l_p; a2.mask = nullptr; a2.o1 = lpT; a2.mode = 2;
  k_ln_gated<<<(BB * NL * NL) / 16, 256, 0, stream>>>(a2);
  ProjArgs a3 = pa; a3.in = i_p; a3.mask = nullptr; a3.o1 = ipb; a3.mode = 3;
  k_ln_gated<<<(BB * NP * NL) / 16, 256, 0, stream>>>(a3);
  ProjArgs a0 = pa; a0.in = z; a0.mask = zm; a0.o1 = ab1T; a0.o2 = ab2T; a0.o3 = ab3; a0.og = gbuf; a0.mode = 0;
  k_ln_gated<<<(BB * NP * NL) / 16, 256, 0, stream>>>(a0);

  // 3) triangle contractions: block = pp@ab1 ; block += ab2@lp^T
  const int gm = NP / 128, gn = NL / 32, batches = BB * C;
  k_bgemm<<<batches * gm * gn, 256, 0, stream>>>(
      ppT, ab1T, blk, NP, NL, NP, NP, NL,
      NP * NP, NP * NL, 0, gm, gn);
  k_bgemm<<<batches * gm * gn, 256, 0, stream>>>(
      ab2T, lpT, blk, NP, NL, NL, NL, NL,
      NP * NL, NL * NL, 1, gm, gn);

  // 4) finish: +ab3*ip, LN(C), @Ws+bs, *g*mask
  k_final<<<(BB * NP * NL) / 16, 256, 0, stream>>>(
      blk, ab3, ipb, gbuf, zm, lnc_w, lnc_b, pWs, bs, out);
}
